// CustomRNNCell_9440338116930
// MI455X (gfx1250) — matrix operand order repaired and checked by reading, not compiled, not run on hardware
//
#include <hip/hip_runtime.h>
#include <hip/hip_bf16.h>

#define B_  64
#define T_  1024
#define D_  256
#define H_  256

typedef __attribute__((ext_vector_type(16))) __bf16 v16bf;
typedef __attribute__((ext_vector_type(8)))  float  v8f;

union AFrag {
    v16bf  v;
    uint4  q[2];
    __bf16 h[16];
};

// ---------------------------------------------------------------------------
// Tensor Data Mover support (guarded: falls back to vector-load staging).
// ---------------------------------------------------------------------------
#if defined(__gfx1250__) && __has_builtin(__builtin_amdgcn_tensor_load_to_lds) && \
    __has_builtin(__builtin_amdgcn_s_wait_tensorcnt)
#define USE_TDM 1
#else
#define USE_TDM 0
#endif

// Split workgroup barrier (signal ... work ... wait) support.
#if defined(__gfx1250__) && __has_builtin(__builtin_amdgcn_s_barrier_signal) && \
    __has_builtin(__builtin_amdgcn_s_barrier_wait)
#define SPLIT_BARRIER 1
#else
#define SPLIT_BARRIER 0
#endif

#if USE_TDM
typedef unsigned int u32x4_t __attribute__((ext_vector_type(4)));
typedef int          i32x4_t __attribute__((ext_vector_type(4)));
typedef int          i32x8_t __attribute__((ext_vector_type(8)));

// Issue a 2D f32 tile load (tile_w x tile_h) from global to LDS via TDM.
// D# layout per CDNA5 ISA 8.3/8.4: group0 = {flags, lds_addr, global_addr,
// type=2}; group1 = {data_size=4B, tensor dims, tile dims, dim0 stride}.
__device__ __forceinline__ void tdm_load_f32_2d(
    unsigned lds_off, const float* gptr,
    unsigned tile_w, unsigned tile_h,
    unsigned tensor_w, unsigned tensor_h, unsigned row_stride_elems)
{
    const unsigned long long ga = (unsigned long long)(uintptr_t)gptr;
    u32x4_t g0;
    g0.x = 1u;                                          // count=1, user mode
    g0.y = lds_off;                                     // LDS byte address
    g0.z = (unsigned)ga;                                // global_addr[31:0]
    g0.w = (unsigned)((ga >> 32) & 0x01FFFFFFull)       // global_addr[56:32]
         | (2u << 30);                                  // type=2 ("image")
    i32x8_t g1;
    g1[0] = (int)(2u << 16);                            // data_size=2 -> 4B
    g1[1] = (int)((tensor_w & 0xFFFFu) << 16);          // tensor_dim0 lo
    g1[2] = (int)((tensor_w >> 16) | ((tensor_h & 0xFFFFu) << 16));
    g1[3] = (int)((tensor_h >> 16) | (tile_w << 16));   // tile_dim0
    g1[4] = (int)(tile_h & 0xFFFFu);                    // tile_dim1 (dim2=0)
    g1[5] = (int)row_stride_elems;                      // dim0_stride[31:0]
    g1[6] = 0;
    g1[7] = 0;
    const i32x4_t z4 = (i32x4_t){0, 0, 0, 0};
#if __clang_major__ >= 23
    const i32x8_t z8 = (i32x8_t){0, 0, 0, 0, 0, 0, 0, 0};
    __builtin_amdgcn_tensor_load_to_lds(g0, g1, z4, z4, z8, 0);
#else
    __builtin_amdgcn_tensor_load_to_lds(g0, g1, z4, z4, 0);
#endif
}
#endif  // USE_TDM

// ---------------------------------------------------------------------------
// Phase 1: xp = x @ Wx + b   (written into d_out's hs region)
// Block = 256 threads = 8 waves; block tile 128M x 128N; wave tile 16M x 128N.
// Wx K-slabs double-buffered into LDS by the TDM (wave 0 issues, all compute),
// then transpose-converted to bf16 fragment layout; B frags via ds_load_b128,
// rotated through two register slots so loads overlap the WMMA chain.
// ---------------------------------------------------------------------------
__global__ __launch_bounds__(256) void xproj_kernel(
    const float* __restrict__ x,    // [B*T, D]
    const float* __restrict__ Wx,   // [D, H]
    const float* __restrict__ bias, // [H]
    float* __restrict__ xp)         // [B*T, H]
{
    // bsh[n][k]: transposed bf16 slab of Wx[k0:k0+32, nb0:nb0+128].
    // Row stride 40 elems = 80B (16B-aligned for ds_load_b128).
    __shared__ __align__(16) __bf16 bsh[128][40];
#if USE_TDM
    __shared__ __align__(16) float raw[2][32][128];     // TDM ping-pong slabs
#endif

    const int tid  = (int)threadIdx.x;
    const int wave = tid >> 5;
    const int lane = tid & 31;
    const int half = lane >> 4;          // 0: lanes 0-15, 1: lanes 16-31
    const int l    = lane & 15;
    const int m0   = (int)blockIdx.x * 128 + wave * 16;
    const int nb0  = (int)blockIdx.y * 128;

    v8f acc[8];
    #pragma unroll
    for (int tt = 0; tt < 8; ++tt) acc[tt] = (v8f){};

    const float* arow = x + (size_t)(m0 + l) * D_;

#if USE_TDM
    // Prime the pipeline: slab for k0=0 -> raw[0].
    if (wave == 0) {
        tdm_load_f32_2d((unsigned)(uintptr_t)&raw[0][0][0], Wx + nb0,
                        /*tile_w=*/128, /*tile_h=*/32,
                        /*tensor_w=*/H_, /*tensor_h=*/D_, /*stride=*/H_);
    }
#endif

    for (int k0 = 0; k0 < D_; k0 += 32) {
#if USE_TDM
        const int cur = (k0 >> 5) & 1;
        if (wave == 0) __builtin_amdgcn_s_wait_tensorcnt(0);  // raw[cur] ready
        __syncthreads();   // publish raw[cur]; bsh free (prev reads done)
        if (wave == 0 && k0 + 32 < D_) {
            tdm_load_f32_2d((unsigned)(uintptr_t)&raw[cur ^ 1][0][0],
                            Wx + (size_t)(k0 + 32) * H_ + nb0,
                            128, 32, H_, D_, H_);      // overlaps compute
        }
        // Transpose-convert raw[cur] (f32 [k][n]) -> bsh (bf16 [n][k]).
        #pragma unroll
        for (int i = 0; i < 16; ++i) {
            const int idx = i * 256 + tid;             // 0 .. 4095
            const int k = idx >> 7, n = idx & 127;
            bsh[n][k] = (__bf16)raw[cur][k][n];
        }
        __syncthreads();
#else
        __syncthreads();   // protect previous iteration's fragment reads
        #pragma unroll
        for (int i = 0; i < 16; ++i) {
            const int idx = i * 256 + tid;
            const int k = idx >> 7, n = idx & 127;
            bsh[n][k] = (__bf16)Wx[(size_t)(k0 + k) * H_ + (nb0 + n)];
        }
        __syncthreads();
#endif

        // A fragment (16x32 bf16): row = m0+l.
        // lanes 0-15: K = k0+{0..7,16..23}; lanes 16-31: K = k0+{8..15,24..31}
        float af[16];
        const float* ap = arow + k0 + half * 8;
        *(float4*)(af + 0)  = ((const float4*)ap)[0];
        *(float4*)(af + 4)  = ((const float4*)ap)[1];
        *(float4*)(af + 8)  = ((const float4*)(ap + 16))[0];
        *(float4*)(af + 12) = ((const float4*)(ap + 16))[1];
        AFrag a;
        #pragma unroll
        for (int i = 0; i < 16; ++i) a.h[i] = (__bf16)af[i];  // v_cvt_pk_bf16

        // 8 N-tiles; B fragment tt+1 is loaded before WMMA tt issues, so the
        // ds_load pair overlaps the matrix op (dscnt<=1 instead of 0).
        AFrag bf[2];
        {
            const uint4* bp = (const uint4*)&bsh[l][half * 8];
            bf[0].q[0] = bp[0];
            bf[0].q[1] = bp[2];
        }
        #pragma unroll
        for (int tt = 0; tt < 8; ++tt) {
            if (tt < 7) {
                const uint4* bp = (const uint4*)&bsh[(tt + 1) * 16 + l][half * 8];
                bf[(tt + 1) & 1].q[0] = bp[0];
                bf[(tt + 1) & 1].q[1] = bp[2];
            }
            acc[tt] = __builtin_amdgcn_wmma_f32_16x16x32_bf16(
                false, a.v, false, bf[tt & 1].v, (short)0, acc[tt], false, false);
        }
    }

    // C layout: VGPR r -> M = r + half*8 ; N = lane&15.
    #pragma unroll
    for (int tt = 0; tt < 8; ++tt) {
        const int col = nb0 + tt * 16 + l;
        const float bv = bias[col];
        #pragma unroll
        for (int r = 0; r < 8; ++r) {
            const int row = m0 + half * 8 + r;
            xp[(size_t)row * H_ + col] = acc[tt][r] + bv;
        }
    }
}

// ---------------------------------------------------------------------------
// Phase 2: sequential scan  h = tanh(xp_t + h @ Wh), in-place over hs.
// 4 blocks (16 batch rows each) x 512 threads (16 waves; wave w owns 16 cols).
// Wh held in registers (8 bf16 B-fragments / wave); h ping-pongs through LDS;
// xp loads pipelined one step ahead; split barrier hides the global h-stores.
// ---------------------------------------------------------------------------
__global__ __launch_bounds__(512) void rnn_scan_kernel(
    float* __restrict__ hs,        // [B, T, H]  in: xp, out: h states
    const float* __restrict__ h0,  // [B, H]
    const float* __restrict__ Wh,  // [H, H]
    float* __restrict__ hlast)     // [B, H]
{
    __shared__ __align__(16) __bf16 hbuf[2][16][H_];  // bf16 ping-pong

    const int wave = (int)(threadIdx.x >> 5);   // 0..15
    const int lane = (int)(threadIdx.x & 31);
    const int half = lane >> 4;
    const int l    = lane & 15;
    const int n0   = wave * 16;                 // this wave's output columns
    const int b0   = (int)blockIdx.x * 16;      // this block's batch rows

    // --- Preload Wh B-fragments into registers (one-time, L2-resident).
    AFrag bw[8];
    #pragma unroll
    for (int kt = 0; kt < 8; ++kt) {
        const float* col = Wh + (size_t)(kt * 32 + half * 8) * H_ + (n0 + l);
        #pragma unroll
        for (int i = 0; i < 16; ++i)
            bw[kt].h[i] = (__bf16)col[(size_t)(i + (i & 8)) * H_];
    }

    // --- Initialize h buffer 0 from h0 (bf16).
    for (int idx = (int)threadIdx.x; idx < 16 * H_; idx += (int)blockDim.x) {
        const int r = idx >> 8, c = idx & (H_ - 1);
        hbuf[0][r][c] = (__bf16)h0[(size_t)(b0 + r) * H_ + c];
    }
    __syncthreads();

    // Per-lane xp/h pointers: row = half*8 + r, col = n0 + l (C layout).
    float* p[8];
    #pragma unroll
    for (int r = 0; r < 8; ++r)
        p[r] = hs + (size_t)(b0 + half * 8 + r) * T_ * H_ + (n0 + l);

    // Prologue: load xp for t=0.
    float xv[8];
    #pragma unroll
    for (int r = 0; r < 8; ++r) xv[r] = p[r][0];

    float lastv[8];
    int cur = 0;

    for (int t = 0; t < T_; ++t) {
        // Issue next step's xp loads now: consumed after step t+1's WMMAs,
        // giving ~a full recurrence step of global-latency slack.
        float xvn[8];
        if (t + 1 < T_) {
            #pragma unroll
            for (int r = 0; r < 8; ++r)
                xvn[r] = p[r][(size_t)(t + 1) * H_];
        }

        // A fragments of h from LDS (row = l, 16B-aligned b128 loads).
        AFrag a[8];
        #pragma unroll
        for (int kt = 0; kt < 8; ++kt) {
            const __bf16* base = &hbuf[cur][l][kt * 32 + half * 8];
            a[kt].q[0] = *(const uint4*)(base);
            a[kt].q[1] = *(const uint4*)(base + 16);
        }

        // h @ Wh : two interleaved accumulators -> dependent chain depth 4.
        v8f acc0 = {}, acc1 = {};
        #pragma unroll
        for (int kt = 0; kt < 8; kt += 2) {
            acc0 = __builtin_amdgcn_wmma_f32_16x16x32_bf16(
                false, a[kt].v,     false, bw[kt].v,     (short)0, acc0, false, false);
            acc1 = __builtin_amdgcn_wmma_f32_16x16x32_bf16(
                false, a[kt + 1].v, false, bw[kt + 1].v, (short)0, acc1, false, false);
        }

        const int nxt = cur ^ 1;
        float hv[8];
        #pragma unroll
        for (int r = 0; r < 8; ++r) {
            hv[r] = tanhf(acc0[r] + acc1[r] + xv[r]);
            hbuf[nxt][half * 8 + r][n0 + l] = (__bf16)hv[r];  // feed next step
            lastv[r] = hv[r];
        }

#if SPLIT_BARRIER
        // Commit LDS stores, signal arrival, then overlap the global stores
        // and register rotation with the other waves' arrival.
        asm volatile("s_wait_dscnt 0x0" ::: "memory");
        __builtin_amdgcn_s_barrier_signal(-1);
        #pragma unroll
        for (int r = 0; r < 8; ++r)
            p[r][(size_t)t * H_] = hv[r];              // overwrite xp with h_t
        #pragma unroll
        for (int r = 0; r < 8; ++r) xv[r] = xvn[r];
        __builtin_amdgcn_s_barrier_wait(-1);
#else
        #pragma unroll
        for (int r = 0; r < 8; ++r)
            p[r][(size_t)t * H_] = hv[r];
        #pragma unroll
        for (int r = 0; r < 8; ++r) xv[r] = xvn[r];
        __syncthreads();
#endif
        cur = nxt;
    }

    #pragma unroll
    for (int r = 0; r < 8; ++r) {
        const int row = half * 8 + r;
        hlast[(size_t)(b0 + row) * H_ + (n0 + l)] = lastv[r];
    }
}

// ---------------------------------------------------------------------------
extern "C" void kernel_launch(void* const* d_in, const int* in_sizes, int n_in,
                              void* d_out, int out_size, void* d_ws, size_t ws_size,
                              hipStream_t stream) {
    const float* x  = (const float*)d_in[0];  // [B,T,D]
    const float* h0 = (const float*)d_in[1];  // [B,H]
    const float* Wx = (const float*)d_in[2];  // [D,H]
    const float* Wh = (const float*)d_in[3];  // [H,H]
    const float* b  = (const float*)d_in[4];  // [H]

    float* hs    = (float*)d_out;                       // [B,T,H]
    float* hlast = hs + (size_t)B_ * T_ * H_;           // [B,H]

    // Phase 1: grid 512 (M) x 2 (N); block tile 128M x 128N.
    xproj_kernel<<<dim3((B_ * T_) / 128, H_ / 128), dim3(256), 0, stream>>>(
        x, Wx, b, hs);

    // Phase 2: one block per 16-row batch tile.
    rnn_scan_kernel<<<dim3(B_ / 16), dim3(512), 0, stream>>>(hs, h0, Wh, hlast);
}
